// EfficientSpectralAttention_76252849373543
// MI455X (gfx1250) — matrix operand order repaired and checked by reading, not compiled, not run on hardware
//
#include <hip/hip_runtime.h>
#include <hip/hip_bf16.h>
#include <math.h>

typedef __attribute__((ext_vector_type(16))) _Float16 v16h;
typedef __attribute__((ext_vector_type(8)))  _Float16 v8h;
typedef __attribute__((ext_vector_type(4)))  _Float16 h4;
typedef __attribute__((ext_vector_type(2)))  _Float16 h2;
typedef __attribute__((ext_vector_type(8)))  float    v8f;

#define CB 4
#define CC 256
#define CH 64
#define CW 64
#define CP (CH*CW)      /* 4096 pixels per image  */
#define NHEADS 8
#define HD 32
#define NGRP 8
#define NT ((size_t)CB*CC*CP)   /* elements per activation tensor */

#define TILE_CO 32
#define TILE_P  256
#define TILE_K  32
#define XROW 40   /* halves per LDS row: 80B pitch -> 16B aligned, bank-friendly */

// ------- gfx1250 async global->LDS path (guarded; fallback = ld/st) -------
// Probe round 2 showed the builtin exists and takes address-space-qualified
// int pointers (AS1 prints as '__device__ int *' in HIP diagnostics).
typedef __attribute__((address_space(1))) int as1_int;
typedef __attribute__((address_space(3))) int as3_int;
#if defined(__AMDGCN__) && __has_builtin(__builtin_amdgcn_global_load_async_to_lds_b32) && __has_builtin(__builtin_amdgcn_s_wait_asynccnt)
#define USE_ASYNC_LDS 1
#else
#define USE_ASYNC_LDS 0
#endif

__device__ __forceinline__ void stage_b32(const float* gsrc, float* ldst) {
#if USE_ASYNC_LDS
  __builtin_amdgcn_global_load_async_to_lds_b32((as1_int*)gsrc, (as3_int*)ldst, 0, 0);
#else
  *ldst = *gsrc;
#endif
}
__device__ __forceinline__ void stage_wait() {
#if USE_ASYNC_LDS
  __builtin_amdgcn_s_wait_asynccnt(0);
#endif
}

// ---------------- wave32 reductions ----------------
__device__ __forceinline__ float waveSum(float v) {
#pragma unroll
  for (int off = 16; off; off >>= 1) v += __shfl_xor(v, off, 32);
  return v;
}
__device__ __forceinline__ float waveMax(float v) {
#pragma unroll
  for (int off = 16; off; off >>= 1) v = fmaxf(v, __shfl_xor(v, off, 32));
  return v;
}
__device__ __forceinline__ float waveMin(float v) {
#pragma unroll
  for (int off = 16; off; off >>= 1) v = fminf(v, __shfl_xor(v, off, 32));
  return v;
}

// ---------------- 1x1 conv as WMMA GEMM ----------------
// Y[b, co, p] = sum_ci W[co,ci] * X[b,ci,p]  (+ bias[co])
// grid: (Cout/32, P/256, B); block: 256 threads = 8 waves (2 co-waves x 4 p-waves)
__global__ __launch_bounds__(256)
void conv1x1_wmma_kernel(const float* __restrict__ X, const float* __restrict__ Wm,
                         const float* __restrict__ bias, float* __restrict__ Y,
                         int Cin) {
  __shared__ alignas(16) _Float16 XsT[TILE_P][XROW];    // pixel-major: B frags contiguous (20 KB)
  __shared__ alignas(16) _Float16 Ws[TILE_CO][XROW];    // co-major:    A frags contiguous (2.5 KB)

  const int tid  = threadIdx.x;
  const int wave = tid >> 5;
  const int lane = tid & 31;
  const int l15  = lane & 15;
  const int hi   = lane >> 4;      // half-wave select
  const int wco  = wave >> 2;      // 0..1 : which 16 c_out rows
  const int wp   = wave & 3;       // 0..3 : which 64 pixels
  const int co0  = blockIdx.x * TILE_CO;
  const int p0   = blockIdx.y * TILE_P;
  const int b    = blockIdx.z;
  const int Cout = gridDim.x * TILE_CO;

  const float* Xb = X + (size_t)b * Cin * CP;

  v8f acc0 = {0,0,0,0,0,0,0,0};
  v8f acc1 = {0,0,0,0,0,0,0,0};
  v8f acc2 = {0,0,0,0,0,0,0,0};
  v8f acc3 = {0,0,0,0,0,0,0,0};

  // per-thread W-tile staging geometry: 4 contiguous k per thread
  const int wr = tid >> 3;             // 0..31  c_out row
  const int wc = (tid & 7) * 4;        // 0,4,..,28 k column

  for (int k0 = 0; k0 < Cin; k0 += TILE_K) {
    __syncthreads();
    // stage X tile transposed: XsT[p_local][k] as f16, pairwise packed stores
#pragma unroll
    for (int k = 0; k < TILE_K; k += 2) {
      const float f0 = Xb[(size_t)(k0 + k) * CP + p0 + tid];
      const float f1 = Xb[(size_t)(k0 + k + 1) * CP + p0 + tid];
      h2 pv = {(_Float16)f0, (_Float16)f1};
      *(h2*)&XsT[tid][k] = pv;
    }
    // stage W tile (32 co x 32 k), 4 halves per thread in one b64 store
    {
      const float* wrow = &Wm[(size_t)(co0 + wr) * Cin + k0 + wc];
      h4 wv4 = {(_Float16)wrow[0], (_Float16)wrow[1], (_Float16)wrow[2], (_Float16)wrow[3]};
      *(h4*)&Ws[wr][wc] = wv4;
    }
    // prefetch next X tile row into L2 (global_prefetch_b8)
    if (k0 + TILE_K < Cin)
      __builtin_prefetch(&Xb[(size_t)(k0 + TILE_K) * CP + p0 + tid], 0, 1);
    __syncthreads();

    // A fragment (16x32): lanes 0-15 hold K0-15, lanes 16-31 hold K16-31 (matches B)
    const int am = wco * 16 + l15;
    v8h a0 = *(const v8h*)&Ws[am][hi * 16];
    v8h a1 = *(const v8h*)&Ws[am][hi * 16 + 8];
    v16h a = __builtin_shufflevector(a0, a1, 0, 1, 2, 3, 4, 5, 6, 7,
                                     8, 9, 10, 11, 12, 13, 14, 15);

    // B fragments (32x16): lanes 0-15 hold K0-15 of column N=lane, lanes 16-31 hold K16-31
    // -> 32 contiguous bytes per lane in XsT: two ds_load_b128
#pragma unroll
    for (int s = 0; s < 4; ++s) {
      const int bn = wp * 64 + s * 16 + l15;
      v8h b0 = *(const v8h*)&XsT[bn][hi * 16];
      v8h b1 = *(const v8h*)&XsT[bn][hi * 16 + 8];
      v16h bf = __builtin_shufflevector(b0, b1, 0, 1, 2, 3, 4, 5, 6, 7,
                                        8, 9, 10, 11, 12, 13, 14, 15);
      v8f* accp = (s == 0) ? &acc0 : (s == 1) ? &acc1 : (s == 2) ? &acc2 : &acc3;
      *accp = __builtin_amdgcn_wmma_f32_16x16x32_f16(
          false, a, false, bf, (short)0, *accp, false, false);
    }
  }

  // D layout: VGPR r, lanes 0-15 -> M=r, lanes 16-31 -> M=8+r; N = lane&15
  float* Yb = Y + (size_t)b * Cout * CP;
  const int sm = (lane < 16) ? 0 : 8;
#pragma unroll
  for (int s = 0; s < 4; ++s) {
    const v8f acc = (s == 0) ? acc0 : (s == 1) ? acc1 : (s == 2) ? acc2 : acc3;
    const int pcol = p0 + wp * 64 + s * 16 + l15;
#pragma unroll
    for (int r = 0; r < 8; ++r) {
      const int co = co0 + wco * 16 + sm + r;
      float v = acc[r];
      if (bias) v += bias[co];
      Yb[(size_t)co * CP + pcol] = v;
    }
  }
}

// ---------------- group-norm statistics ----------------
// one block per (b, group); group = 32 consecutive channels => 131072 contiguous floats
__global__ __launch_bounds__(256)
void gn_stats_kernel(const float* __restrict__ X, float* __restrict__ mean,
                     float* __restrict__ invstd) {
  const int bg = blockIdx.x;                       // b*NGRP + g
  const float* base = X + (size_t)bg * (CC / NGRP) * CP;
  const int n = (CC / NGRP) * CP;                  // 131072
  float s = 0.f, ss = 0.f;
  for (int i = threadIdx.x; i < n; i += 256) {
    float v = base[i];
    s += v; ss += v * v;
  }
  s = waveSum(s); ss = waveSum(ss);
  __shared__ float sh[2][8];
  const int wave = threadIdx.x >> 5, lane = threadIdx.x & 31;
  if (lane == 0) { sh[0][wave] = s; sh[1][wave] = ss; }
  __syncthreads();
  if (threadIdx.x == 0) {
    float ts = 0.f, tss = 0.f;
#pragma unroll
    for (int i = 0; i < 8; ++i) { ts += sh[0][i]; tss += sh[1][i]; }
    const float m = ts / (float)n;
    const float var = tss / (float)n - m * m;
    mean[bg] = m;
    invstd[bg] = rsqrtf(var + 1e-5f);
  }
}

// ---------------- per-token rank-1 softmax attention ----------------
// one block per (b, head, 32 pixels): stage 32x32 q/k/v tiles in LDS (coalesced,
// async-to-LDS when available), then one wave per token, lane = head_dim index.
__global__ __launch_bounds__(256)
void attention_kernel(const float* __restrict__ q_pre, const float* __restrict__ k_pre,
                      const float* __restrict__ v_in,
                      const float* __restrict__ mq, const float* __restrict__ iq,
                      const float* __restrict__ mk, const float* __restrict__ ik,
                      const float* __restrict__ gqg, const float* __restrict__ gqb,
                      const float* __restrict__ gkg, const float* __restrict__ gkb,
                      float* __restrict__ out) {
  __shared__ float Qs[32][33];
  __shared__ float Ks[32][33];
  __shared__ float Vs[32][33];
  const int tid = threadIdx.x;
  const int nptiles = CP / 32;                       // 128
  const int pt   = blockIdx.x % nptiles;
  const int head = (blockIdx.x / nptiles) % NHEADS;
  const int b    = blockIdx.x / (nptiles * NHEADS);
  const size_t base = ((size_t)b * CC + head * HD) * CP + pt * 32;

  // coalesced staging of three 32(ch) x 32(px) tiles
#pragma unroll
  for (int r = 0; r < 4; ++r) {
    const int idx = r * 256 + tid;
    const int cl = idx >> 5, pl = idx & 31;
    const size_t off = base + (size_t)cl * CP + pl;
    stage_b32(q_pre + off, &Qs[cl][pl]);
    stage_b32(k_pre + off, &Ks[cl][pl]);
    stage_b32(v_in + off, &Vs[cl][pl]);
  }
  stage_wait();
  __syncthreads();

  const int wave = tid >> 5, lane = tid & 31;
  const int c  = head * HD + lane;                   // head == group
  const int bg = b * NGRP + head;
  const float mqv = mq[bg], iqv = iq[bg], mkv = mk[bg], ikv = ik[bg];
  const float gqgv = gqg[c], gqbv = gqb[c], gkgv = gkg[c], gkbv = gkb[c];
  const float scale = 0.17677669529663687f;          // 1/sqrt(32)

  // each wave owns 4 pixel columns; no cross-wave column sharing
#pragma unroll
  for (int t = 0; t < 4; ++t) {
    const int pl = wave * 4 + t;
    float q = (Qs[lane][pl] - mqv) * iqv * gqgv + gqbv;
    float k = (Ks[lane][pl] - mkv) * ikv * gkgv + gkbv;
    q *= __frcp_rn(fmaxf(sqrtf(waveSum(q * q)), 1e-12f));
    k *= __frcp_rn(fmaxf(sqrtf(waveSum(k * k)), 1e-12f));
    Ks[lane][pl] = k;                                // normalized k, same-wave column
    const float kmax = waveMax(k), kmin = waveMin(k);
    const float rowmax = scale * ((q >= 0.f) ? q * kmax : q * kmin);
    float num = 0.f, den = 0.f;
#pragma unroll 8
    for (int e = 0; e < 32; ++e) {
      const float ke = Ks[e][pl];                    // LDS broadcast read
      const float ve = Vs[e][pl];
      const float tt = __expf(q * ke * scale - rowmax);
      den += tt;
      num = fmaf(tt, ve, num);
    }
    Qs[lane][pl] = num / den;                        // reuse Q tile for output
  }
  __syncthreads();
#pragma unroll
  for (int r = 0; r < 4; ++r) {
    const int idx = r * 256 + tid;
    const int cl = idx >> 5, pl = idx & 31;
    out[base + (size_t)cl * CP + pl] = Qs[cl][pl];
  }
}

// ---------------- depthwise 3x3 (+GELU) (+residual add) ----------------
__global__ __launch_bounds__(256)
void dwconv3x3_kernel(const float* __restrict__ X, const float* __restrict__ Wd,
                      const float* __restrict__ bd, const float* __restrict__ addend,
                      float* __restrict__ Y, int apply_gelu) {
  const size_t i = (size_t)blockIdx.x * 256 + threadIdx.x;  // over B*C*H*W
  const int wcol = (int)(i % CW);
  const int hrow = (int)((i / CW) % CH);
  const int c = (int)((i / (CH * CW)) % CC);
  const float* plane = X + (i - (size_t)hrow * CW - wcol);
  const float* wk = Wd + c * 9;
  float acc = bd[c];
#pragma unroll
  for (int dy = -1; dy <= 1; ++dy) {
    const int y = hrow + dy;
    if (y < 0 || y >= CH) continue;
#pragma unroll
    for (int dx = -1; dx <= 1; ++dx) {
      const int x = wcol + dx;
      if (x < 0 || x >= CW) continue;
      acc = fmaf(wk[(dy + 1) * 3 + (dx + 1)], plane[y * CW + x], acc);
    }
  }
  if (apply_gelu) acc = 0.5f * acc * (1.f + erff(acc * 0.70710678118654752f));
  if (addend) acc += addend[i];
  Y[i] = acc;
}

// ---------------- group-norm apply (in place) ----------------
__global__ __launch_bounds__(256)
void gn_apply_kernel(float* __restrict__ X, const float* __restrict__ mean,
                     const float* __restrict__ invstd, const float* __restrict__ gamma,
                     const float* __restrict__ beta) {
  const size_t i = (size_t)blockIdx.x * 256 + threadIdx.x;
  const int c = (int)((i / CP) % CC);
  const int b = (int)(i / ((size_t)CC * CP));
  const int bg = b * NGRP + c / (CC / NGRP);
  X[i] = (X[i] - mean[bg]) * invstd[bg] * gamma[c] + beta[c];
}

extern "C" void kernel_launch(void* const* d_in, const int* in_sizes, int n_in,
                              void* d_out, int out_size, void* d_ws, size_t ws_size,
                              hipStream_t stream) {
  const float* x        = (const float*)d_in[0];
  const float* wq       = (const float*)d_in[1];
  const float* gq_gamma = (const float*)d_in[2];
  const float* gq_beta  = (const float*)d_in[3];
  const float* wk       = (const float*)d_in[4];
  const float* gk_gamma = (const float*)d_in[5];
  const float* gk_beta  = (const float*)d_in[6];
  const float* wv       = (const float*)d_in[7];
  const float* wp       = (const float*)d_in[8];
  const float* bp       = (const float*)d_in[9];
  const float* gp_gamma = (const float*)d_in[10];
  const float* gp_beta  = (const float*)d_in[11];
  const float* dw1      = (const float*)d_in[12];
  const float* db1      = (const float*)d_in[13];
  const float* dw2      = (const float*)d_in[14];
  const float* db2      = (const float*)d_in[15];
  float* out = (float*)d_out;

  float* ws   = (float*)d_ws;
  float* q_pre = ws;                 // [B,C,P]
  float* k_pre = ws + NT;            // [B,C,P]
  float* v_buf = ws + 2 * NT;        // [B,C,P]
  float* acc   = ws + 3 * NT;        // attn out, then attn+pos
  float* stats = ws + 4 * NT;        // 6 x 32 floats
  float* mq = stats, *iq = stats + 32, *mk = stats + 64, *ik = stats + 96;
  float* mp = stats + 128, *ip = stats + 160;

  const dim3 ggrid(CC / TILE_CO, CP / TILE_P, CB);       // (8,16,4)
  const int nelem_blocks = (int)(NT / 256);              // 16384

  // q/k/v pointwise convs (WMMA)
  conv1x1_wmma_kernel<<<ggrid, 256, 0, stream>>>(x, wq, nullptr, q_pre, CC);
  conv1x1_wmma_kernel<<<ggrid, 256, 0, stream>>>(x, wk, nullptr, k_pre, CC);
  conv1x1_wmma_kernel<<<ggrid, 256, 0, stream>>>(x, wv, nullptr, v_buf, CC);

  // group-norm stats for q and k
  gn_stats_kernel<<<CB * NGRP, 256, 0, stream>>>(q_pre, mq, iq);
  gn_stats_kernel<<<CB * NGRP, 256, 0, stream>>>(k_pre, mk, ik);

  // rank-1 softmax attention, LDS-tiled
  attention_kernel<<<CB * NHEADS * (CP / 32), 256, 0, stream>>>(
      q_pre, k_pre, v_buf, mq, iq, mk, ik,
      gq_gamma, gq_beta, gk_gamma, gk_beta, acc);

  // positional branch: dw3x3 + GELU -> dw3x3, added onto attention output
  // (q_pre buffer is dead after attention; reuse for the intermediate)
  dwconv3x3_kernel<<<nelem_blocks, 256, 0, stream>>>(x, dw1, db1, nullptr, q_pre, 1);
  dwconv3x3_kernel<<<nelem_blocks, 256, 0, stream>>>(q_pre, dw2, db2, acc, acc, 0);

  // projection conv (WMMA, fused bias) straight into d_out
  conv1x1_wmma_kernel<<<ggrid, 256, 0, stream>>>(acc, wp, bp, out, CC);

  // final group-norm, in place on d_out
  gn_stats_kernel<<<CB * NGRP, 256, 0, stream>>>(out, mp, ip);
  gn_apply_kernel<<<nelem_blocks, 256, 0, stream>>>(out, mp, ip, gp_gamma, gp_beta);
}
